// MultiHeadSlotSelfAttn_10075993277018
// MI455X (gfx1250) — matrix operand order repaired and checked by reading, not compiled, not run on hardware
//
#include <hip/hip_runtime.h>
#include <hip/hip_bf16.h>

typedef __attribute__((ext_vector_type(16))) _Float16 v16h;
typedef __attribute__((ext_vector_type(8)))  _Float16 v8h;
typedef __attribute__((ext_vector_type(8)))  float    v8f;
typedef __attribute__((ext_vector_type(4)))  float    v4f;
typedef __attribute__((ext_vector_type(4)))  unsigned u32x4;
typedef __attribute__((ext_vector_type(8)))  unsigned u32x8;

#define DIM      1024
#define HEADS    16
#define HEAD_DIM 64
#define SEQ      2048
#define BATCH    2
#define MROWS    (BATCH * SEQ)   /* 4096 */
#define SM_SCALE 0.125f          /* 1/sqrt(64) */

static __device__ __forceinline__ v8f wmma16x16x32(v16h a, v16h b, v8f c) {
    // D = A(16x32 f16) * B(32x16 f16) + C(16x16 f32)
    return __builtin_amdgcn_wmma_f32_16x16x32_f16(false, a, false, b, (short)0, c,
                                                  false, false);
}

static __device__ __forceinline__ v16h cat8(v8h lo, v8h hi) {
    v16h r;
#pragma unroll
    for (int i = 0; i < 8; ++i) { r[i] = lo[i]; r[i + 8] = hi[i]; }
    return r;
}

// gfx1250 async DMA: global -> LDS, 16 bytes per lane, tracked by ASYNCcnt.
// VDST carries the wave-relative LDS byte offset (= low 32 bits of the flat
// shared-aperture address per the ISA's flat->LDS mapping).
static __device__ __forceinline__ void async_copy16(void* lds, const void* glb) {
    asm volatile("global_load_async_to_lds_b128 %0, %1, off"
                 :: "v"((unsigned)(size_t)lds), "v"(glb)
                 : "memory");
}
static __device__ __forceinline__ void async_wait0() {
    asm volatile("s_wait_asynccnt 0x0" ::: "memory");
}

// ---------------------------------------------------------------------------
// Tensor Data Mover: one descriptor-driven DMA moves a 2D tile (tile_w x
// tile_h, 2-byte elements, row stride `stride_elems`) from global memory into
// LDS at `lds_off`, packed rows (tile rows land contiguously -> matches our
// LDS tile layouts exactly). Tracked by TENSORcnt. D# built per ISA 8.3/8.4:
//   group0: count=1 | lds_addr | global_addr[56:0] | type=2
//   group1: data_size=1(2B) | tensor_dim0/1 | tile_dim0/1 | dim0_stride
// ---------------------------------------------------------------------------
static __device__ __forceinline__ void tdm_load_2d(unsigned lds_off, const void* gaddr,
                                                   unsigned tile_w, unsigned tile_h,
                                                   unsigned tensor_w, unsigned tensor_h,
                                                   unsigned stride_elems) {
    const unsigned long long ga = (unsigned long long)gaddr;
    u32x4 g0;
    g0[0] = 1u;                                             // count=1, user D#
    g0[1] = lds_off;                                        // lds_addr (bytes)
    g0[2] = (unsigned)(ga & 0xffffffffu);                   // global_addr[31:0]
    g0[3] = (unsigned)((ga >> 32) & 0x1ffffffu) | (2u << 30); // addr[56:32]|type=2
    u32x8 g1;
    g1[0] = 1u << 16;                                       // data_size=1 (2B), wg_mask=0
    g1[1] = (tensor_w & 0xffffu) << 16;                     // tensor_dim0[15:0]
    g1[2] = ((tensor_w >> 16) & 0xffffu) | ((tensor_h & 0xffffu) << 16);
    g1[3] = ((tensor_h >> 16) & 0xffffu) | (tile_w << 16);  // tile_dim0
    g1[4] = tile_h & 0xffffu;                               // tile_dim1 (tile_dim2=0)
    g1[5] = stride_elems;                                   // tensor_dim0_stride[31:0]
    g1[6] = 0u;                                             // stride hi, dim1_stride lo
    g1[7] = 0u;
    asm volatile("tensor_load_to_lds %0, %1" :: "s"(g0), "s"(g1) : "memory");
}
static __device__ __forceinline__ void tensor_wait0() {
    __builtin_amdgcn_s_wait_tensorcnt((unsigned short)0);
}

// ---------------------------------------------------------------------------
// Stage 0: fp32 -> f16 conversion (vectorized, n is a multiple of 8)
// ---------------------------------------------------------------------------
__global__ void __launch_bounds__(256)
cvt_f32_to_f16(const float* __restrict__ x, _Float16* __restrict__ y, int n) {
    int i = (blockIdx.x * 256 + threadIdx.x) * 8;
    if (i >= n) return;
    v4f a = *(const v4f*)(x + i);
    v4f b = *(const v4f*)(x + i + 4);
    v8h o;
#pragma unroll
    for (int k = 0; k < 4; ++k) { o[k] = (_Float16)a[k]; o[k + 4] = (_Float16)b[k]; }
    *(v8h*)(y + i) = o;
}

// ---------------------------------------------------------------------------
// Tiled GEMM: Y[M,N] = X[M,K] * W[N,K]^T + bias, M=4096, N=K=1024.
// Block: 128 threads (4 waves), 64x64 output tile, K-step 32.
// Double-buffered LDS tiles filled by the Tensor Data Mover (wave 0 issues
// two descriptor DMAs per K-slice); ONE barrier per K-step.
// OUT_MODE 0: f16 output scattered into [B,H,S,Hd] (QKV head layout)
// OUT_MODE 1: f32 output row-major [M,N] (final projection)
// ---------------------------------------------------------------------------
template <int OUT_MODE>
__global__ void __launch_bounds__(128)
gemm_wmma_kernel(const _Float16* __restrict__ X,
                 const _Float16* __restrict__ W,
                 const float* __restrict__ bias,
                 void* __restrict__ Yout)
{
    __shared__ _Float16 As[2][64][32];
    __shared__ _Float16 Bs[2][64][32];

    const int tid   = threadIdx.x;
    const int lane  = tid & 31;
    const int wave  = tid >> 5;          // 0..3
    const int mtile = blockIdx.y * 64;
    const int ntile = blockIdx.x * 64;

    // ISA fragment indexing (16-bit A 16x32; 16x16 f32 C/D)
    const int arow  = wave * 16 + (lane & 15);
    const int khalf = (lane >> 4) * 8;   // A-fragment K sub-chunk
    const int kb16  = (lane >> 4) * 16;  // unused; B-fragment uses khalf like A

    const _Float16* xtile = X + (size_t)mtile * DIM;   // 64 rows starting here
    const _Float16* wtile = W + (size_t)ntile * DIM;

    v8f acc[4] = {};

    // Preload K-slice 0 into buffer 0 via TDM (tile 32 wide x 64 rows)
    if (wave == 0) {
        tdm_load_2d((unsigned)(size_t)&As[0][0][0], xtile,     32, 64, DIM, MROWS, DIM);
        tdm_load_2d((unsigned)(size_t)&Bs[0][0][0], wtile,     32, 64, DIM, DIM,   DIM);
    }

    for (int k0 = 0; k0 < DIM; k0 += 32) {
        const int p = (k0 >> 5) & 1;
        if (wave == 0) tensor_wait0();   // buffer p's DMA has landed
        __syncthreads();                 // all waves done reading p^1, safe to refill

        if (k0 + 32 < DIM) {             // kick off next K-slice into p^1
            if (wave == 0) {
                tdm_load_2d((unsigned)(size_t)&As[p ^ 1][0][0], xtile + k0 + 32,
                            32, 64, DIM, MROWS, DIM);
                tdm_load_2d((unsigned)(size_t)&Bs[p ^ 1][0][0], wtile + k0 + 32,
                            32, 64, DIM, DIM, DIM);
            }
            __builtin_prefetch(xtile + (size_t)arow * DIM + k0 + 64, 0, 3);
            __builtin_prefetch(wtile + (size_t)arow * DIM + k0 + 64, 0, 3);
        }

        v16h a = cat8(*(const v8h*)&As[p][arow][khalf],
                      *(const v8h*)&As[p][arow][16 + khalf]);
#pragma unroll
        for (int nt = 0; nt < 4; ++nt) {
            v16h b = cat8(*(const v8h*)&Bs[p][nt * 16 + (lane & 15)][khalf],
                          *(const v8h*)&Bs[p][nt * 16 + (lane & 15)][16 + khalf]);
            acc[nt] = wmma16x16x32(a, b, acc[nt]);
        }
    }

    const int nloc  = lane & 15;
    const int rbase = (lane >> 4) * 8;
#pragma unroll
    for (int nt = 0; nt < 4; ++nt) {
        const int n  = ntile + nt * 16 + nloc;
        const float bv = bias[n];
#pragma unroll
        for (int r = 0; r < 8; ++r) {
            const int   m = mtile + wave * 16 + rbase + r;
            const float v = acc[nt][r] + bv;
            if (OUT_MODE == 0) {
                const int bb = m >> 11, ss = m & (SEQ - 1);
                const int hh = n >> 6,  dd = n & (HEAD_DIM - 1);
                ((_Float16*)Yout)[(((size_t)bb * HEADS + hh) * SEQ + ss) * HEAD_DIM + dd] =
                    (_Float16)v;
            } else {
                ((float*)Yout)[(size_t)m * DIM + n] = v;
            }
        }
    }
}

// ---------------------------------------------------------------------------
// Fused flash attention. Grid: (S/64, B*H). Block: 128 threads (4 waves).
// Each block owns 64 query rows of one (b,h); each wave owns 16 of them.
// Q tile arrives via a TDM descriptor DMA; K tiles via per-lane async
// global->LDS; V is transposed through VGPRs. Online softmax in fp32;
// P staged through LDS (C-layout -> A-layout).
// ---------------------------------------------------------------------------
__global__ void __launch_bounds__(128)
flash_attn_kernel(const _Float16* __restrict__ Q,
                  const _Float16* __restrict__ K,
                  const _Float16* __restrict__ V,
                  _Float16* __restrict__ O)   // [B*S, D] f16, head-merged
{
    __shared__ _Float16 Qs[64][64];
    __shared__ _Float16 Ks[64][64];
    __shared__ _Float16 VTs[64][64];       // transposed: VTs[d][k]
    __shared__ _Float16 Ps[4][16][64];     // per-wave P tile (q x k)

    const int tid  = threadIdx.x;
    const int lane = tid & 31;
    const int wave = tid >> 5;
    const int bh    = blockIdx.y;          // b*H + h
    const int qtile = blockIdx.x * 64;
    const size_t base = (size_t)bh * SEQ * HEAD_DIM;

    const int ldRow = tid >> 1;            // 0..63
    const int ldOff = (tid & 1) * 32;      // 0 or 32

    // Load Q tile (64 x 64) via Tensor Data Mover
    if (wave == 0) {
        tdm_load_2d((unsigned)(size_t)&Qs[0][0],
                    Q + base + (size_t)qtile * HEAD_DIM,
                    HEAD_DIM, 64, HEAD_DIM, SEQ, HEAD_DIM);
        tensor_wait0();
    }
    __syncthreads();

    const int khalf = (lane >> 4) * 8;
    const int kb16  = (lane >> 4) * 16;
    const int arow  = wave * 16 + (lane & 15);

    // Q A-fragments for d-chunks [0,32) and [32,64)
    v16h qf[2];
#pragma unroll
    for (int ks = 0; ks < 2; ++ks)
        qf[ks] = cat8(*(const v8h*)&Qs[arow][ks * 32 + khalf],
                      *(const v8h*)&Qs[arow][ks * 32 + 16 + khalf]);

    float mrun[8], lsum[8];
    v8f acc[4] = {};
#pragma unroll
    for (int r = 0; r < 8; ++r) { mrun[r] = -1e30f; lsum[r] = 0.f; }

    for (int j = 0; j < SEQ; j += 64) {
        __syncthreads();   // previous iteration's reads of Ks/VTs are done
        {
            const _Float16* ksrc = K + base + (size_t)(j + ldRow) * HEAD_DIM + ldOff;
            const _Float16* vsrc = V + base + (size_t)(j + ldRow) * HEAD_DIM + ldOff;
#pragma unroll
            for (int c = 0; c < 4; ++c)
                async_copy16(&Ks[ldRow][ldOff + c * 8], ksrc + c * 8);
#pragma unroll
            for (int c = 0; c < 4; ++c) {
                v8h vv = *(const v8h*)(vsrc + c * 8);
#pragma unroll
                for (int e = 0; e < 8; ++e)
                    VTs[ldOff + c * 8 + e][ldRow] = vv[e];   // transpose store
            }
        }
        async_wait0();
        __syncthreads();

        // S = Q * K^T  (16 q-rows x 64 k-cols per wave)
        v8f sacc[4] = {};
#pragma unroll
        for (int kt = 0; kt < 4; ++kt) {
            const int brow = kt * 16 + (lane & 15);          // key row = score col
            v16h b0 = cat8(*(const v8h*)&Ks[brow][khalf], *(const v8h*)&Ks[brow][16 + khalf]);         // d in [0,32)
            v16h b1 = cat8(*(const v8h*)&Ks[brow][32 + khalf], *(const v8h*)&Ks[brow][48 + khalf]);    // d in [32,64)
            sacc[kt] = wmma16x16x32(qf[0], b0, sacc[kt]);
            sacc[kt] = wmma16x16x32(qf[1], b1, sacc[kt]);
        }
#pragma unroll
        for (int kt = 0; kt < 4; ++kt)
#pragma unroll
            for (int r = 0; r < 8; ++r)
                sacc[kt][r] *= SM_SCALE;

        // Row max across 4 k-tiles then across the 16 lanes holding the row
        float mloc[8];
#pragma unroll
        for (int r = 0; r < 8; ++r) {
            float mv = sacc[0][r];
#pragma unroll
            for (int kt = 1; kt < 4; ++kt) mv = fmaxf(mv, sacc[kt][r]);
#pragma unroll
            for (int off = 1; off < 16; off <<= 1)
                mv = fmaxf(mv, __shfl_xor(mv, off, 32));
            mloc[r] = mv;
        }
        float mnew[8], alpha[8], rsum[8];
#pragma unroll
        for (int r = 0; r < 8; ++r) {
            mnew[r]  = fmaxf(mrun[r], mloc[r]);
            alpha[r] = __expf(mrun[r] - mnew[r]);
            mrun[r]  = mnew[r];
            rsum[r]  = 0.f;
        }

        // P = exp(S - m); stage to LDS for the A-layout reload
        const int prow = (lane >> 4) * 8;
        const int pcol = lane & 15;
#pragma unroll
        for (int kt = 0; kt < 4; ++kt)
#pragma unroll
            for (int r = 0; r < 8; ++r) {
                float p = __expf(sacc[kt][r] - mnew[r]);
                rsum[r] += p;
                Ps[wave][prow + r][kt * 16 + pcol] = (_Float16)p;
            }
#pragma unroll
        for (int r = 0; r < 8; ++r) {
#pragma unroll
            for (int off = 1; off < 16; off <<= 1)
                rsum[r] += __shfl_xor(rsum[r], off, 32);
            lsum[r] = lsum[r] * alpha[r] + rsum[r];
        }
#pragma unroll
        for (int dt = 0; dt < 4; ++dt)
#pragma unroll
            for (int r = 0; r < 8; ++r)
                acc[dt][r] *= alpha[r];

        // O += P * V    (intra-wave Ps reuse: DS ops are in-order per wave)
        const int parow = lane & 15;
#pragma unroll
        for (int ks = 0; ks < 2; ++ks) {
            v16h pa = cat8(*(const v8h*)&Ps[wave][parow][ks * 32 + khalf],
                           *(const v8h*)&Ps[wave][parow][ks * 32 + 16 + khalf]);
#pragma unroll
            for (int dt = 0; dt < 4; ++dt) {
                v16h bv = cat8(*(const v8h*)&VTs[dt * 16 + (lane & 15)][ks * 32 + khalf],
                               *(const v8h*)&VTs[dt * 16 + (lane & 15)][ks * 32 + 16 + khalf]);
                acc[dt] = wmma16x16x32(pa, bv, acc[dt]);
            }
        }
    }

    // Normalize and write to [B*S, D] f16 (heads merged back)
    const int bb = bh >> 4, hh = bh & 15;
#pragma unroll
    for (int r = 0; r < 8; ++r) {
        const int   q   = qtile + wave * 16 + (lane >> 4) * 8 + r;
        const float inv = 1.0f / lsum[r];
#pragma unroll
        for (int dt = 0; dt < 4; ++dt) {
            const int d = dt * 16 + (lane & 15);
            O[((size_t)bb * SEQ + q) * DIM + hh * HEAD_DIM + d] =
                (_Float16)(acc[dt][r] * inv);
        }
    }
}

// ---------------------------------------------------------------------------
// Host launcher
// ---------------------------------------------------------------------------
extern "C" void kernel_launch(void* const* d_in, const int* in_sizes, int n_in,
                              void* d_out, int out_size, void* d_ws, size_t ws_size,
                              hipStream_t stream) {
    const float* Z  = (const float*)d_in[0];
    const float* Wq = (const float*)d_in[1];
    const float* bq = (const float*)d_in[2];
    const float* Wk = (const float*)d_in[3];
    const float* bk = (const float*)d_in[4];
    const float* Wv = (const float*)d_in[5];
    const float* bv = (const float*)d_in[6];
    const float* Wo = (const float*)d_in[7];
    const float* bo = (const float*)d_in[8];

    const size_t nZ = (size_t)MROWS * DIM;   // 4,194,304
    const size_t nW = (size_t)DIM * DIM;     // 1,048,576

    // Workspace layout (f16): Zh | Wqh | Wkh | Wvh | Woh | Qh | Kh | Vh | Ah  (~48 MB)
    _Float16* Zh  = (_Float16*)d_ws;
    _Float16* Wqh = Zh  + nZ;
    _Float16* Wkh = Wqh + nW;
    _Float16* Wvh = Wkh + nW;
    _Float16* Woh = Wvh + nW;
    _Float16* Qh  = Woh + nW;   // [B,H,S,Hd]
    _Float16* Kh  = Qh  + nZ;
    _Float16* Vh  = Kh  + nZ;
    _Float16* Ah  = Vh  + nZ;   // [B*S, D]

    cvt_f32_to_f16<<<dim3((unsigned)(nZ / 2048)), 256, 0, stream>>>(Z,  Zh,  (int)nZ);
    cvt_f32_to_f16<<<dim3((unsigned)(nW / 2048)), 256, 0, stream>>>(Wq, Wqh, (int)nW);
    cvt_f32_to_f16<<<dim3((unsigned)(nW / 2048)), 256, 0, stream>>>(Wk, Wkh, (int)nW);
    cvt_f32_to_f16<<<dim3((unsigned)(nW / 2048)), 256, 0, stream>>>(Wv, Wvh, (int)nW);
    cvt_f32_to_f16<<<dim3((unsigned)(nW / 2048)), 256, 0, stream>>>(Wo, Woh, (int)nW);

    dim3 gg(DIM / 64, MROWS / 64);   // (16, 64)
    gemm_wmma_kernel<0><<<gg, 128, 0, stream>>>(Zh, Wqh, bq, Qh);
    gemm_wmma_kernel<0><<<gg, 128, 0, stream>>>(Zh, Wkh, bk, Kh);
    gemm_wmma_kernel<0><<<gg, 128, 0, stream>>>(Zh, Wvh, bv, Vh);

    flash_attn_kernel<<<dim3(SEQ / 64, BATCH * HEADS), 128, 0, stream>>>(Qh, Kh, Vh, Ah);

    gemm_wmma_kernel<1><<<gg, 128, 0, stream>>>(Ah, Woh, bo, d_out);
}
